// MixtureOfExperts_88665304859102
// MI455X (gfx1250) — matrix operand order repaired and checked by reading, not compiled, not run on hardware
//
#include <hip/hip_runtime.h>

typedef __attribute__((ext_vector_type(16))) __bf16 v16bf;
typedef __attribute__((ext_vector_type(8)))  float  v8f;
typedef __attribute__((address_space(3)))    __bf16 lds_bf16;

#define B_TOK 2048
#define DDIM  1024
#define HDIM  2048
#define ODIM  1024
#define NEXP  32
#define KSEL  8
#define MT    128   // M tile (pairs)
#define NT    128   // N tile
#define KT    32    // K step (bf16 wmma depth)
#define ASTRIDE 40  // A LDS row stride (32 + 8 pad)
#define BSTRIDE 136 // B LDS row stride (128 + 8 pad)

union U16bf { uint4 u[2]; v16bf v; };
union U8bf  { __bf16 h[8]; uint4 u; };
struct bf4  { __bf16 a, b, c, d; };

__device__ __forceinline__ unsigned lds_off(const __bf16* p) {
    return (unsigned)(unsigned long long)(lds_bf16*)p;   // addrspacecast -> LDS byte offset
}

__device__ __forceinline__ void async_copy_a32(unsigned lds_dst, const __bf16* gsrc) {
    unsigned long long ga = (unsigned long long)gsrc;
    // copies 32B (16 bf16): offset applies to BOTH global and LDS sides (ISA 15.18)
    asm volatile("global_load_async_to_lds_b128 %0, %1, off"
                 :: "v"(lds_dst), "v"(ga) : "memory");
    asm volatile("global_load_async_to_lds_b128 %0, %1, off offset:16"
                 :: "v"(lds_dst), "v"(ga) : "memory");
}

__device__ __forceinline__ void wait_async0() {
    asm volatile("s_wait_asynccnt 0x0" ::: "memory");
}

// ---------------- x -> bf16 ----------------
__global__ void moe_cvt_x(const float* __restrict__ x, __bf16* __restrict__ xb, int n4) {
    int i = blockIdx.x * blockDim.x + threadIdx.x;
    if (i < n4) {
        float4 f = reinterpret_cast<const float4*>(x)[i];
        bf4 o;
        o.a = (__bf16)f.x; o.b = (__bf16)f.y; o.c = (__bf16)f.z; o.d = (__bf16)f.w;
        reinterpret_cast<bf4*>(xb)[i] = o;
    }
}

// ---------------- gating + top-8 (1 wave per token, lane == expert) ----------------
__global__ void moe_gate_topk(const float* __restrict__ x, const float* __restrict__ Wg,
                              const float* __restrict__ bg, int* __restrict__ topi,
                              float* __restrict__ topw) {
    const int b = blockIdx.x;
    const int lane = threadIdx.x;  // 0..31 == expert id (wave32)
    const float4* xr = reinterpret_cast<const float4*>(x  + (size_t)b    * DDIM);
    const float4* wr = reinterpret_cast<const float4*>(Wg + (size_t)lane * DDIM);
    float acc = 0.f;
    #pragma unroll 4
    for (int d = 0; d < DDIM / 4; ++d) {
        float4 xv = xr[d], wv = wr[d];
        acc += xv.x * wv.x + xv.y * wv.y + xv.z * wv.z + xv.w * wv.w;
    }
    acc += bg[lane];

    float m = acc;
    for (int off = 16; off > 0; off >>= 1) m = fmaxf(m, __shfl_xor(m, off, 32));
    float g = __expf(acc - m);
    float s = g;
    for (int off = 16; off > 0; off >>= 1) s += __shfl_xor(s, off, 32);
    float gate = g / s;

    float v = gate, myw = 0.f, wsum = 0.f;
    #pragma unroll
    for (int k = 0; k < KSEL; ++k) {
        float bv = v; int bi = lane;
        for (int off = 16; off > 0; off >>= 1) {
            float ov = __shfl_xor(bv, off, 32);
            int   oi = __shfl_xor(bi, off, 32);
            if (ov > bv || (ov == bv && oi < bi)) { bv = ov; bi = oi; }
        }
        wsum += bv;
        if (lane == k)  { myw = bv; topi[b * KSEL + k] = bi; }
        if (lane == bi) v = -1.f;
    }
    if (lane < KSEL) topw[b * KSEL + lane] = myw / wsum;
}

// ---------------- route pairs to per-expert lists ----------------
__global__ void moe_scatter(const int* __restrict__ topi, int* __restrict__ counts,
                            int* __restrict__ perm) {
    int p = blockIdx.x * blockDim.x + threadIdx.x;
    if (p < B_TOK * KSEL) {
        int e = topi[p];
        int slot = atomicAdd(&counts[e], 1);   // slot order irrelevant: one row per pair id
        perm[e * B_TOK + slot] = p;
    }
}

// ---------------- grouped GEMM: MTxNT tile, 8 waves, double-buffered LDS ----------------
template <bool LAYER1>
__global__ __launch_bounds__(256, 2) void moe_gemm(
    const __bf16* __restrict__ Abase, const float* __restrict__ Wall,
    const float* __restrict__ biasAll, const int* __restrict__ perm,
    const int* __restrict__ counts, void* __restrict__ outp,
    int kdim, int n)
{
    const int e     = blockIdx.z;
    const int cnt   = counts[e];
    const int mtile = blockIdx.y;
    if (mtile * MT >= cnt) return;                  // uniform early exit
    const int n0   = blockIdx.x * NT;
    const int tid  = threadIdx.x;
    const int wv   = tid >> 5;
    const int lane = tid & 31;
    const int hf   = lane >> 4;
    const int lm   = lane & 15;
    const int* eperm = perm + e * B_TOK;

    __shared__ __bf16 As[2][MT * ASTRIDE];
    __shared__ __bf16 Bs[2][KT * BSTRIDE];

    const float* Wk   = Wall    + (size_t)e * kdim * n;
    const float* bias = biasAll + (size_t)e * n;

    // A-tile loader assignment: 2 threads per row, 32B (16 bf16) each, via async-to-LDS
    const int ar = tid >> 1;
    const int ah = tid & 1;
    const int aidx = mtile * MT + ar;
    const int ap = (aidx < cnt) ? eperm[aidx] : -1;
    const __bf16* arow =
        (ap >= 0) ? (LAYER1 ? (Abase + (size_t)(ap >> 3) * kdim)   // pair -> token row of xb
                            : (Abase + (size_t)ap * kdim))         // pair row of Hbuf
                  : Abase;                                         // padding: harmless row 0

    // B-tile loader assignment: 16 fp32 -> 16 bf16 per thread
    const int kk  = tid >> 3;
    const int seg = tid & 7;
    const float4* bsrc0 = reinterpret_cast<const float4*>(Wk + (size_t)kk * n + n0 + seg * 16);
    const size_t  bsrcStep = (size_t)KT * n / 4;    // float4 step per K tile

    // accumulators, bias-initialized (C layout: VGPR r -> M=r+8*hf, col = lm)
    v8f acc[8];
    #pragma unroll
    for (int s2 = 0; s2 < 8; ++s2) {
        float bv = bias[n0 + s2 * 16 + lm];
        #pragma unroll
        for (int r = 0; r < 8; ++r) acc[s2][r] = bv;
    }

    auto loadBregs = [&](int it, float4 f[4]) {
        const float4* bs = bsrc0 + (size_t)it * bsrcStep;
        #pragma unroll
        for (int q = 0; q < 4; ++q) f[q] = bs[q];
    };
    auto commitB = [&](__bf16* Bb, const float4 f[4]) {
        #pragma unroll
        for (int q = 0; q < 2; ++q) {
            U8bf pk;
            pk.h[0] = (__bf16)f[q*2].x;   pk.h[1] = (__bf16)f[q*2].y;
            pk.h[2] = (__bf16)f[q*2].z;   pk.h[3] = (__bf16)f[q*2].w;
            pk.h[4] = (__bf16)f[q*2+1].x; pk.h[5] = (__bf16)f[q*2+1].y;
            pk.h[6] = (__bf16)f[q*2+1].z; pk.h[7] = (__bf16)f[q*2+1].w;
            reinterpret_cast<uint4*>(Bb + kk * BSTRIDE + seg * 16)[q] = pk.u;
        }
    };
    // 4-deep software-pipelined fragment rotation: each WMMA consumes a fragment
    // loaded up to 3 fragments (48 scalar bf16 LDS reads) earlier, enabling partial s_wait_dscnt + wmma overlap.
    auto computeTile = [&](const __bf16* Ab, const __bf16* Bb) {
        U16bf af;
        const uint4* pa =
            reinterpret_cast<const uint4*>(Ab + (wv * 16 + lm) * ASTRIDE + hf * 8);
        af.u[0] = pa[0];
        af.u[1] = pa[2];   // +16 elements
        const __bf16* brow = Bb + lm;
        U16bf f0, f1, f2, f3;
#define LOADB(dst, s) do { \
        _Pragma("unroll") for (int j = 0; j < 16; ++j) \
            dst.v[j] = brow[((j & 8) * 2 + hf * 8 + (j & 7)) * BSTRIDE + (s) * 16]; } while (0)
#define WMMA8(i, fr) acc[i] = __builtin_amdgcn_wmma_f32_16x16x32_bf16( \
        false, af.v, false, fr.v, (short)0, acc[i], false, false)
        LOADB(f0, 0); LOADB(f1, 1); LOADB(f2, 2); LOADB(f3, 3);
        WMMA8(0, f0); LOADB(f0, 4);
        WMMA8(1, f1); LOADB(f1, 5);
        WMMA8(2, f2); LOADB(f2, 6);
        WMMA8(3, f3); LOADB(f3, 7);
        WMMA8(4, f0);
        WMMA8(5, f1);
        WMMA8(6, f2);
        WMMA8(7, f3);
#undef LOADB
#undef WMMA8
    };

    const int nIter = kdim / KT;    // 32 or 64: always even

    // ---- prologue: stage tile 0 into buffer 0 ----
    async_copy_a32(lds_off(&As[0][ar * ASTRIDE + ah * 16]), arow + ah * 16);
    {
        float4 f[4];
        loadBregs(0, f);
        commitB(Bs[0], f);
    }
    wait_async0();
    __syncthreads();

    for (int it = 0; it < nIter; it += 2) {
        // ---- phase 0: compute buffer 0, stage (it+1) into buffer 1 ----
        {
            const bool hn = (it + 1) < nIter;
            if (hn)
                async_copy_a32(lds_off(&As[1][ar * ASTRIDE + ah * 16]),
                               arow + (it + 1) * KT + ah * 16);
            float4 f[4];
            if (hn) loadBregs(it + 1, f);
            computeTile(As[0], Bs[0]);
            if (hn) commitB(Bs[1], f);
            wait_async0();
            __syncthreads();
        }
        if (it + 1 >= nIter) break;
        // ---- phase 1: compute buffer 1, stage (it+2) into buffer 0 ----
        {
            const bool hn = (it + 2) < nIter;
            if (hn)
                async_copy_a32(lds_off(&As[0][ar * ASTRIDE + ah * 16]),
                               arow + (it + 2) * KT + ah * 16);
            float4 f[4];
            if (hn) loadBregs(it + 2, f);
            computeTile(As[1], Bs[1]);
            if (hn) commitB(Bs[0], f);
            wait_async0();
            __syncthreads();
        }
    }

    // ---- epilogue: per-lane rows m = r + 8*hf of this wave's 16-row strip ----
    int prow[8];
    #pragma unroll
    for (int r = 0; r < 8; ++r) {
        int idx = mtile * MT + wv * 16 + (r + hf * 8);
        prow[r] = (idx < cnt) ? eperm[idx] : -1;
    }
    #pragma unroll
    for (int s2 = 0; s2 < 8; ++s2) {
        int col = n0 + s2 * 16 + lm;
        #pragma unroll
        for (int r = 0; r < 8; ++r) {
            if (prow[r] < 0) continue;
            float vv = acc[s2][r];
            if (LAYER1) {
                ((__bf16*)outp)[(size_t)prow[r] * n + col] = (__bf16)fmaxf(vv, 0.f);
            } else {
                ((float*)outp)[(size_t)prow[r] * n + col] = vv;
            }
        }
    }
}

// ---------------- weighted combine ----------------
__global__ void moe_combine(const float* __restrict__ Y, const float* __restrict__ topw,
                            float* __restrict__ out) {
    int b = blockIdx.x;
    float w[KSEL];
    #pragma unroll
    for (int k = 0; k < KSEL; ++k) w[k] = topw[b * KSEL + k];
    for (int o = threadIdx.x; o < ODIM; o += blockDim.x) {
        float acc = 0.f;
        #pragma unroll
        for (int k = 0; k < KSEL; ++k)
            acc += w[k] * Y[(size_t)(b * KSEL + k) * ODIM + o];
        out[(size_t)b * ODIM + o] = acc;
    }
}

extern "C" void kernel_launch(void* const* d_in, const int* in_sizes, int n_in,
                              void* d_out, int out_size, void* d_ws, size_t ws_size,
                              hipStream_t stream) {
    (void)in_sizes; (void)n_in; (void)out_size; (void)ws_size;
    const float* x  = (const float*)d_in[0];
    const float* Wg = (const float*)d_in[1];
    const float* bg = (const float*)d_in[2];
    const float* W1 = (const float*)d_in[3];
    const float* b1 = (const float*)d_in[4];
    const float* W2 = (const float*)d_in[5];
    const float* b2 = (const float*)d_in[6];
    float* out = (float*)d_out;

    char* ws = (char*)d_ws;
    size_t off = 0;
    auto alloc = [&](size_t bytes) {
        void* p = ws + off;
        off += (bytes + 255) & ~(size_t)255;
        return p;
    };
    int*    topi   = (int*)   alloc((size_t)B_TOK * KSEL * 4);
    float*  topw   = (float*) alloc((size_t)B_TOK * KSEL * 4);
    int*    counts = (int*)   alloc((size_t)NEXP * 4);
    int*    perm   = (int*)   alloc((size_t)NEXP * B_TOK * 4);
    __bf16* xb     = (__bf16*)alloc((size_t)B_TOK * DDIM * 2);
    __bf16* Hbuf   = (__bf16*)alloc((size_t)B_TOK * KSEL * HDIM * 2);
    float*  Ybuf   = (float*) alloc((size_t)B_TOK * KSEL * ODIM * 4);

    hipMemsetAsync(counts, 0, NEXP * sizeof(int), stream);

    moe_cvt_x<<<(B_TOK * DDIM / 4 + 255) / 256, 256, 0, stream>>>(x, xb, B_TOK * DDIM / 4);
    moe_gate_topk<<<B_TOK, 32, 0, stream>>>(x, Wg, bg, topi, topw);
    moe_scatter<<<(B_TOK * KSEL) / 256, 256, 0, stream>>>(topi, counts, perm);

    dim3 g1(HDIM / NT, B_TOK / MT, NEXP);
    moe_gemm<true><<<g1, 256, 0, stream>>>(xb, W1, b1, perm, counts, (void*)Hbuf, DDIM, HDIM);

    dim3 g2(ODIM / NT, B_TOK / MT, NEXP);
    moe_gemm<false><<<g2, 256, 0, stream>>>(Hbuf, W2, b2, perm, counts, (void*)Ybuf, HDIM, ODIM);

    moe_combine<<<B_TOK, 256, 0, stream>>>(Ybuf, topw, out);
}
